// VisionMambaBackbone_84911503442223
// MI455X (gfx1250) — matrix operand order repaired and checked by reading, not compiled, not run on hardware
//
#include <hip/hip_runtime.h>
#include <stdint.h>
#include <math.h>

#define CDIV(a,b) (((a)+(b)-1)/(b))

// ---------------------------------------------------------------------------
// CDNA5 WMMA types (wave32). 16x16x32 bf16 -> f32 accumulate.
// ---------------------------------------------------------------------------
typedef __attribute__((ext_vector_type(16))) __bf16 v16bf;
typedef __attribute__((ext_vector_type(8)))  float  v8f;

union FragBF { v16bf v; __bf16 b[16]; unsigned short s[16]; uint32_t u[8]; };

__device__ __forceinline__ float siluf(float x) { return x / (1.0f + __expf(-x)); }

// ---------------------------------------------------------------------------
// Pack fp32 (possibly strided) weight -> bf16 row-major [N][K]
// (native hardware f32->bf16 convert, RNE)
// ---------------------------------------------------------------------------
__global__ void k_pack_bf16(const float* __restrict__ src, unsigned short* __restrict__ dst,
                            int N, int K, long rowStride, long colStride) {
  long i = (long)blockIdx.x * blockDim.x + threadIdx.x;
  if (i >= (long)N * (long)K) return;
  int n = (int)(i / K), k = (int)(i % K);
  __bf16 h = (__bf16)(src[(long)n * rowStride + (long)k * colStride]);
  dst[i] = __builtin_bit_cast(unsigned short, h);
}

// ---------------------------------------------------------------------------
// GEMM: C[M,N] (+)= act( A[M,K](fp32, ld=lda) * W[N,K]^T (bf16) + bias )
// Wave computes 16(M) x 64(N); block = 4 waves -> 64x64 tile.
// rowShift: shift A rows by +s within length-rowsPerSeg segments (zero pad) —
//           used to express the k=3 embedding conv as 3 accumulated GEMMs.
// actMode: 0=none, 1=softplus.  M%64==0, N%64==0, K%32==0 guaranteed here.
// ---------------------------------------------------------------------------
__global__ __launch_bounds__(128)
void k_gemm_bf16w(const float* __restrict__ A, long lda,
                  const unsigned short* __restrict__ W,
                  float* __restrict__ C,
                  int M, int N, int K,
                  int rowShift, int rowsPerSeg,
                  const float* __restrict__ bias,
                  int actMode, int accumulate) {
  const int lane = threadIdx.x & 31;
  const int wave = threadIdx.x >> 5;
  const int n0 = blockIdx.x * 64;
  const int m0 = blockIdx.y * 64 + wave * 16;
  const int rl = lane & 15;     // row-within-tile / col-within-tile
  const int hi = lane >> 4;     // 0: low half lanes, 1: high half lanes

  const int r = m0 + rl;
  int sr = r; bool rvalid = true;
  if (rowShift != 0) {
    int seg = r / rowsPerSeg;
    int off = r - seg * rowsPerSeg + rowShift;
    rvalid = (off >= 0) && (off < rowsPerSeg);
    sr = rvalid ? (seg * rowsPerSeg + off) : r;   // clamped-safe source row
  }
  const float mask = rvalid ? 1.0f : 0.0f;        // branch-free zero padding

  v8f acc[4];
  #pragma unroll
  for (int j = 0; j < 4; ++j)
    #pragma unroll
    for (int v = 0; v < 8; ++v) acc[j][v] = 0.0f;

  for (int kk = 0; kk < K; kk += 32) {
    // A fragment: 16x32 bf16. lane<16: row rl, K {kk..kk+15};
    //             lane>=16: row rl, K {kk+16..kk+31}.
    FragBF af;
    const float* pa = A + (long)sr * lda + kk + (hi ? 16 : 0);
    #pragma unroll
    for (int i = 0; i < 8; ++i) af.b[i]     = (__bf16)(pa[i]      * mask);
    #pragma unroll
    for (int i = 0; i < 8; ++i) af.b[8 + i] = (__bf16)(pa[8 + i] * mask);
    __builtin_prefetch((const void*)(pa + 32), 0, 1);   // next K slice

    #pragma unroll
    for (int j = 0; j < 4; ++j) {
      // B fragment: 32x16 bf16. lane n: col n0+j*16+rl, K lo/hi 16 per hi bit.
      FragBF bfr;
      const int col = n0 + j * 16 + rl;
      const uint32_t* pb = (const uint32_t*)(W + (long)col * K + kk + (hi ? 16 : 0));
      #pragma unroll
      for (int i = 0; i < 8; ++i) bfr.u[i] = pb[i];
      __builtin_prefetch((const void*)(pb + 16), 0, 1);
      acc[j] = __builtin_amdgcn_wmma_f32_16x16x32_bf16(
          false, af.v, false, bfr.v, (short)0, acc[j], false, false);
    }
  }

  // Epilogue. C/D layout: lane&15 = N col; lane>=16 -> rows M+8..M+15.
  const int mb = m0 + hi * 8;
  #pragma unroll
  for (int j = 0; j < 4; ++j) {
    const int nn = n0 + j * 16 + rl;
    const float bv = bias ? bias[nn] : 0.0f;
    #pragma unroll
    for (int v = 0; v < 8; ++v) {
      const int row = mb + v;
      long idx = (long)row * N + nn;
      float val = acc[j][v];
      if (accumulate) val += C[idx];
      val += bv;
      if (actMode == 1) val = (val > 30.0f) ? val : log1pf(__expf(val));
      C[idx] = val;
    }
  }
}

// ---------------------------------------------------------------------------
// LayerNorm over C=512 (one wave per row, in-place capable), optional ReLU.
// ---------------------------------------------------------------------------
__global__ __launch_bounds__(256)
void k_layernorm512(const float* __restrict__ X, float* __restrict__ Y,
                    const float* __restrict__ g, const float* __restrict__ b,
                    int R, int doRelu) {
  const int wave = threadIdx.x >> 5, lane = threadIdx.x & 31;
  const int row = blockIdx.x * 8 + wave;
  if (row >= R) return;
  const float* xr = X + (long)row * 512;
  float v[16], s = 0.0f, s2 = 0.0f;
  #pragma unroll
  for (int i = 0; i < 16; ++i) {
    v[i] = xr[i * 32 + lane];
    s += v[i]; s2 += v[i] * v[i];
  }
  #pragma unroll
  for (int o = 16; o >= 1; o >>= 1) {
    s  += __shfl_xor(s,  o, 32);
    s2 += __shfl_xor(s2, o, 32);
  }
  const float mean = s * (1.0f / 512.0f);
  const float var  = s2 * (1.0f / 512.0f) - mean * mean;
  const float inv  = rsqrtf(var + 1e-5f);
  float* yr = Y + (long)row * 512;
  #pragma unroll
  for (int i = 0; i < 16; ++i) {
    const int c = i * 32 + lane;
    float o = (v[i] - mean) * inv * g[c] + b[c];
    if (doRelu) o = fmaxf(o, 0.0f);
    yr[c] = o;
  }
}

// ---------------------------------------------------------------------------
// Causal depthwise conv (k=4, pad 3) + SiLU over xp = xz[:, :, 0:1024].
// rev=1: operate on the L-reversed sequence; output stays in "logical" order.
// ---------------------------------------------------------------------------
__global__ void k_dwconv_silu(const float* __restrict__ XZ,  // R x 2048
                              const float* __restrict__ w,   // 1024 x 4
                              const float* __restrict__ bias,
                              float* __restrict__ XC,        // R x 1024
                              int Bn, int L, int rev) {
  long i = (long)blockIdx.x * blockDim.x + threadIdx.x;
  long total = (long)Bn * L * 1024;
  if (i >= total) return;
  const int d = (int)(i & 1023);
  long t = i >> 10;
  const int l = (int)(t % L);
  const int b = (int)(t / L);
  float acc = bias[d];
  #pragma unroll
  for (int tt = 0; tt < 4; ++tt) {
    const int lp = l - 3 + tt;
    if (lp >= 0) {
      const int ar = rev ? (L - 1 - lp) : lp;
      acc += w[d * 4 + tt] * XZ[((long)b * L + ar) * 2048 + d];
    }
  }
  XC[((long)b * L + l) * 1024 + d] = siluf(acc);
}

// ---------------------------------------------------------------------------
// Selective scan. Thread owns one (b,d) channel, 16-state recurrence in regs.
// B/C rows (shared across all d) staged in LDS per 64-step chunk.
// Y = scan(h) + xc * D.
// ---------------------------------------------------------------------------
__global__ __launch_bounds__(256)
void k_selective_scan(const float* __restrict__ XC,   // R x 1024
                      const float* __restrict__ DT,   // R x 1024 (post-softplus)
                      const float* __restrict__ XDB,  // R x 64 (B: 32..47, C: 48..63)
                      const float* __restrict__ A_log,// 1024 x 16
                      const float* __restrict__ Dp,   // 1024
                      float* __restrict__ Y,          // R x 1024
                      int L) {
  __shared__ float sB[64][16];
  __shared__ float sC[64][16];
  const int tid = threadIdx.x;
  const int bb = blockIdx.x >> 2;
  const int d  = ((blockIdx.x & 3) << 8) + tid;
  float a[16], h[16];
  #pragma unroll
  for (int n = 0; n < 16; ++n) { a[n] = -__expf(A_log[d * 16 + n]); h[n] = 0.0f; }
  const float Dd = Dp[d];
  const long rowBase = (long)bb * L;
  for (int l0 = 0; l0 < L; l0 += 64) {
    #pragma unroll
    for (int i = 0; i < 4; ++i) {
      const int f = tid + (i << 8);
      const int ll = f >> 4, n = f & 15;
      const long src = (rowBase + l0 + ll) * 64;
      sB[ll][n] = XDB[src + 32 + n];
      sC[ll][n] = XDB[src + 48 + n];
    }
    __syncthreads();
    for (int ll = 0; ll < 64; ++ll) {
      const long r = rowBase + l0 + ll;
      const float dt = DT[r * 1024 + d];
      const float xv = XC[r * 1024 + d];
      const float dx = dt * xv;
      float y = 0.0f;
      #pragma unroll
      for (int n = 0; n < 16; ++n) {
        const float dA = __expf(dt * a[n]);
        h[n] = dA * h[n] + dx * sB[ll][n];
        y += h[n] * sC[ll][n];
      }
      Y[r * 1024 + d] = y + xv * Dd;
    }
    __syncthreads();
  }
}

// ---------------------------------------------------------------------------
// out[l] = (yf[l] + yb[L-1-l]) * silu(z[l])   (z = xz cols 1024..2047)
// ---------------------------------------------------------------------------
__global__ void k_combine_gate(const float* __restrict__ YF, const float* __restrict__ YB,
                               const float* __restrict__ XZ, float* __restrict__ OUT,
                               int Bn, int L) {
  long i = (long)blockIdx.x * blockDim.x + threadIdx.x;
  long total = (long)Bn * L * 1024;
  if (i >= total) return;
  const int d = (int)(i & 1023);
  long t = i >> 10;
  const int l = (int)(t % L);
  const int b = (int)(t / L);
  const long rf = (long)b * L + l;
  const long rb = (long)b * L + (L - 1 - l);
  const float z = XZ[rf * 2048 + 1024 + d];
  OUT[rf * 1024 + d] = (YF[rf * 1024 + d] + YB[rb * 1024 + d]) * siluf(z);
}

// ---------------------------------------------------------------------------
// Misc elementwise
// ---------------------------------------------------------------------------
__global__ void k_copy_or_add(float* __restrict__ dst, const float* __restrict__ src,
                              long n, int addFlag) {
  long i = (long)blockIdx.x * blockDim.x + threadIdx.x;
  if (i >= n) return;
  dst[i] = addFlag ? (dst[i] + src[i]) : src[i];
}

__global__ void k_downsample2(float* __restrict__ dst, const float* __restrict__ src,
                              int Bn, int L2, int C) {
  long i = (long)blockIdx.x * blockDim.x + threadIdx.x;
  long total = (long)Bn * L2 * C;
  if (i >= total) return;
  const int c = (int)(i % C);
  long t = i / C;
  const int l = (int)(t % L2);
  const int b = (int)(t / L2);
  dst[i] = src[(((long)b * L2 * 2) + 2 * l) * C + c];
}

// feats are (B, 512, L): dst[b,c,l] = src[(b*L+l)*512 + c]
__global__ void k_store_feat(float* __restrict__ dst, const float* __restrict__ src,
                             int Bn, int L) {
  long i = (long)blockIdx.x * blockDim.x + threadIdx.x;
  long total = (long)Bn * 512 * L;
  if (i >= total) return;
  const int l = (int)(i % L);
  long t = i / L;
  const int c = (int)(t % 512);
  const int b = (int)(t / 512);
  dst[i] = src[((long)b * L + l) * 512 + c];
}

// x (B, C, L) -> XT (B*L, C)
__global__ void k_transpose_in(const float* __restrict__ x, float* __restrict__ XT,
                               int Bn, int C, int L) {
  long i = (long)blockIdx.x * blockDim.x + threadIdx.x;
  long total = (long)Bn * L * C;
  if (i >= total) return;
  const int c = (int)(i % C);
  long t = i / C;
  const int l = (int)(t % L);
  const int b = (int)(t / L);
  XT[i] = x[((long)b * C + c) * L + l];
}

__global__ void k_fill_ones(float* __restrict__ dst, long n) {
  long i = (long)blockIdx.x * blockDim.x + threadIdx.x;
  if (i < n) dst[i] = 1.0f;
}

// ---------------------------------------------------------------------------
// Host orchestration
// ---------------------------------------------------------------------------
extern "C" void kernel_launch(void* const* d_in, const int* in_sizes, int n_in,
                              void* d_out, int out_size, void* d_ws, size_t ws_size,
                              hipStream_t stream) {
  (void)in_sizes; (void)out_size; (void)ws_size;
  if (n_in < 135) return;

  const int B = 2, L0 = 1024;

  // d_in layout (jax sorted-key tree-flatten):
  //   [0..89]   branch blocks b0..b4, 18 tensors each
  //   [90..97]  embd0 {b, be, g, w}, embd1 {b, be, g, w}
  //   [98..133] stem blocks s0..s1, 18 tensors each
  //   [134]     x
  // per block: 0 ln_b, 1 ln_g, 2 A_log, 3 A_log_b, 4 D, 5 D_b, 6 W_dt, 7 W_dt_b,
  //            8 W_in, 9 W_out, 10 W_x, 11 W_x_b, 12 b_dt, 13 b_dt_b,
  //            14 conv_b, 15 conv_b_b, 16 conv_w, 17 conv_w_b

  // ---- workspace layout (max R = B*L0 = 2048 rows) ----
  unsigned short* wpack = (unsigned short*)d_ws;              // 2048*512 bf16 = 2 MB
  float* F = (float*)((char*)d_ws + (size_t)2 * 1024 * 1024);
  float* xz  = F; F += (size_t)2048 * 2048;
  float* bh  = F; F += (size_t)2048 * 512;
  float* brs = F; F += (size_t)2048 * 512;
  float* bh2 = F; F += (size_t)2048 * 512;
  float* brs2= F; F += (size_t)2048 * 512;
  float* lnb = F; F += (size_t)2048 * 512;
  float* xc  = F; F += (size_t)2048 * 1024;
  float* dtb = F; F += (size_t)2048 * 1024;
  float* yf  = F; F += (size_t)2048 * 1024;
  float* yb  = F; F += (size_t)2048 * 1024;
  float* xdb = F; F += (size_t)2048 * 64;

  float* H = bh;  float* RES = brs;  float* HB = bh2;  float* RESB = brs2;
  float* out = (float*)d_out;

  auto run_gemm = [&](const float* A, long lda, const float* Wf, int N, int K,
                      float* C, int M, int rowShift, int rowsPerSeg,
                      const float* bias, int act, int accum,
                      long wRowStride, long wColStride) {
    long total = (long)N * K;
    k_pack_bf16<<<CDIV(total, 256), 256, 0, stream>>>(Wf, wpack, N, K, wRowStride, wColStride);
    dim3 g(N / 64, M / 64);
    k_gemm_bf16w<<<g, 128, 0, stream>>>(A, lda, wpack, C, M, N, K,
                                        rowShift, rowsPerSeg, bias, act, accum);
  };

  auto run_block = [&](void* const* bp, int L, bool first) {
    const int R = B * L;
    const float* ln_b  = (const float*)bp[0];
    const float* ln_g  = (const float*)bp[1];
    const float* W_in  = (const float*)bp[8];
    const float* W_out = (const float*)bp[9];

    // residual update: res = h (first) / res + h
    k_copy_or_add<<<CDIV((long)R * 512, 256), 256, 0, stream>>>(RES, H, (long)R * 512, first ? 0 : 1);
    // LN(res) -> lnb
    k_layernorm512<<<CDIV(R, 8), 256, 0, stream>>>(RES, lnb, ln_g, ln_b, R, 0);
    // xz = lnb @ W_in.T   (R x 2048)
    run_gemm(lnb, 512, W_in, 2048, 512, xz, R, 0, R, nullptr, 0, 0, 512, 1);

    for (int dir = 0; dir < 2; ++dir) {
      const float* A_log = (const float*)bp[2 + dir];
      const float* Dp    = (const float*)bp[4 + dir];
      const float* W_dt  = (const float*)bp[6 + dir];
      const float* W_x   = (const float*)bp[10 + dir];
      const float* b_dt  = (const float*)bp[12 + dir];
      const float* convb = (const float*)bp[14 + dir];
      const float* convw = (const float*)bp[16 + dir];
      float* ybuf = dir ? yb : yf;

      k_dwconv_silu<<<CDIV((long)R * 1024, 256), 256, 0, stream>>>(xz, convw, convb, xc, B, L, dir);
      // xdb = xc @ W_x.T   (R x 64)
      run_gemm(xc, 1024, W_x, 64, 1024, xdb, R, 0, R, nullptr, 0, 0, 1024, 1);
      // dt = softplus(xdb[:, :32] @ W_dt.T + b_dt)  (R x 1024)
      run_gemm(xdb, 64, W_dt, 1024, 32, dtb, R, 0, R, b_dt, 1, 0, 32, 1);
      // selective scan (+ xc*D)
      k_selective_scan<<<B * 4, 256, 0, stream>>>(xc, dtb, xdb, A_log, Dp, ybuf, L);
    }

    // (yf + rev(yb)) * silu(z) -> xc (reuse)
    k_combine_gate<<<CDIV((long)R * 1024, 256), 256, 0, stream>>>(yf, yb, xz, xc, B, L);
    // h_new = comb @ W_out.T  (R x 512)
    run_gemm(xc, 1024, W_out, 512, 1024, H, R, 0, R, nullptr, 0, 0, 1024, 1);
  };

  // ===================== embedding =====================
  const float* xin = (const float*)d_in[134];
  k_transpose_in<<<CDIV((long)B * L0 * 512, 256), 256, 0, stream>>>(xin, H, B, 512, L0);

  auto run_embd = [&](int base, const float* Ain, float* Cout) {
    const float* eb  = (const float*)d_in[base + 0];
    const float* ebe = (const float*)d_in[base + 1];
    const float* eg  = (const float*)d_in[base + 2];
    const float* ew  = (const float*)d_in[base + 3];   // (512, 512, 3)
    for (int t = 0; t < 3; ++t) {
      // slice w[:,:,t]: rowStride 512*3, colStride 3; row shift t-1 (pad 1)
      run_gemm(Ain, 512, ew + t, 512, 512, Cout, B * L0, t - 1, L0,
               (t == 2) ? eb : nullptr, 0, (t == 0) ? 0 : 1, (long)512 * 3, 3);
    }
    k_layernorm512<<<CDIV(B * L0, 8), 256, 0, stream>>>(Cout, Cout, eg, ebe, B * L0, 1);
  };
  run_embd(90, H, RES);   // layer 1: H(=x^T) -> RES
  run_embd(94, RES, H);   // layer 2: RES -> H

  // ===================== stem =====================
  run_block(d_in + 98, L0, true);
  run_block(d_in + 116, L0, false);

  size_t featOff = 0;
  k_store_feat<<<CDIV((long)B * 512 * L0, 256), 256, 0, stream>>>(out + featOff, H, B, L0);
  featOff += (size_t)B * 512 * L0;

  // ===================== branch =====================
  int L = L0;
  for (int i = 0; i < 5; ++i) {
    run_block(d_in + i * 18, L, false);
    const int L2 = L / 2;
    k_downsample2<<<CDIV((long)B * L2 * 512, 256), 256, 0, stream>>>(HB, H, B, L2, 512);
    k_downsample2<<<CDIV((long)B * L2 * 512, 256), 256, 0, stream>>>(RESB, RES, B, L2, 512);
    float* t0 = H;   H = HB;     HB = t0;
    float* t1 = RES; RES = RESB; RESB = t1;
    L = L2;
    k_store_feat<<<CDIV((long)B * 512 * L, 256), 256, 0, stream>>>(out + featOff, H, B, L);
    featOff += (size_t)B * 512 * L;
  }

  // ===================== masks (all ones) =====================
  const long maskElems = (long)B * (1024 + 512 + 256 + 128 + 64 + 32);
  k_fill_ones<<<CDIV(maskElems, 256), 256, 0, stream>>>(out + featOff, maskElems);
}
